// NerModel_67396626808876
// MI455X (gfx1250) — matrix operand order repaired and checked by reading, compiled, not run on hardware
//
#include <hip/hip_runtime.h>
#include <hip/hip_bf16.h>
#include <stdint.h>

typedef __bf16 bf16_t;
typedef __bf16 v16bf __attribute__((ext_vector_type(16)));
typedef float  v8f   __attribute__((ext_vector_type(8)));
typedef unsigned int v4u __attribute__((ext_vector_type(4)));
typedef int v4i __attribute__((ext_vector_type(4)));
typedef int v8i __attribute__((ext_vector_type(8)));

#if defined(__has_builtin)
#  if __has_builtin(__builtin_amdgcn_tensor_load_to_lds) && __has_builtin(__builtin_amdgcn_s_wait_tensorcnt)
#    define USE_TDM 1
#  endif
#endif
#ifndef USE_TDM
#  define USE_TDM 0
#endif

constexpr int kB = 64, kT = 256, kE = 300, kD = 600, kDp = 608;  // Dp = 19*32
constexpr int kH = 512, kG = 2048, kL = 9;
constexpr int LOGITS_OFF = 0;
constexpr int LENS_OFF = kB * kT * kL;   // 147456
constexpr int LL_OFF   = LENS_OFF + kB;  // 147520
constexpr int TAGS_OFF = LL_OFF + kB;    // 147584

__device__ __forceinline__ float sigmoidf_(float x) { return 1.0f / (1.0f + __expf(-x)); }

__device__ __forceinline__ v8f wmma_bf16(v16bf a, v16bf b, v8f c) {
  // D = A(16x32 bf16) * B(32x16 bf16) + C(16x16 f32)
  return __builtin_amdgcn_wmma_f32_16x16x32_bf16(false, a, false, b, (short)0, c, false, false);
}

__device__ __forceinline__ float wave_sum(float v) {
#pragma unroll
  for (int off = 16; off; off >>= 1) v += __shfl_xor(v, off, 32);
  return v;
}

#if USE_TDM
// TDM: DMA a 2D bf16 tile (tile_d1 rows x tile_d0 contiguous elems, row stride
// stride0 elems) from global into LDS (contiguous). Descriptor packed per
// CDNA5 ISA Sec 8.3/8.4. data_size = 2 bytes.
__device__ __forceinline__ void tdm_load_2d_bf16(const void* gaddr, void* lds_ptr,
                                                 unsigned tile_d0, unsigned tile_d1,
                                                 unsigned tensor_d0, unsigned tensor_d1,
                                                 unsigned long long stride0) {
  const unsigned lds_off = (unsigned)(size_t)lds_ptr;  // LDS aperture: low 32 bits
  const unsigned long long ga = (unsigned long long)(size_t)gaddr;
  v4u g0;
  g0[0] = 1u;                                   // count=1 (valid), user descriptor
  g0[1] = lds_off;                              // [63:32] lds_addr
  g0[2] = (unsigned)(ga & 0xffffffffu);         // [95:64] global_addr lo
  g0[3] = (unsigned)((ga >> 32) & 0x01ffffffu)  // [120:96] global_addr hi
          | (2u << 30);                         // [127:126] type = 2 ("image")
  v8i g1;
  g1[0] = (int)(1u << 16);                                        // data_size=1 -> 2B
  g1[1] = (int)((tensor_d0 & 0xffffu) << 16);                     // tensor_dim0[15:0]
  g1[2] = (int)((tensor_d0 >> 16) | ((tensor_d1 & 0xffffu) << 16));
  g1[3] = (int)((tensor_d1 >> 16) | (tile_d0 << 16));             // tile_dim0
  g1[4] = (int)(tile_d1 & 0xffffu);                               // tile_dim1 (tile_dim2=0)
  g1[5] = (int)(unsigned)(stride0 & 0xffffffffu);                 // dim0_stride lo
  g1[6] = (int)(unsigned)((stride0 >> 32) & 0xffffu);             // dim0_stride hi
  g1[7] = 0;
  v4i gz = {0, 0, 0, 0};
#if defined(__clang_major__) && (__clang_major__ >= 23)
  v8i gz8 = {0, 0, 0, 0, 0, 0, 0, 0};
  __builtin_amdgcn_tensor_load_to_lds(g0, g1, gz, gz, gz8, 0);
#else
  __builtin_amdgcn_tensor_load_to_lds(g0, g1, gz, gz, 0);
#endif
}
#endif

// ---------------- utility kernels ----------------

__global__ void k_zero_f32(float* p, int n) {
  int i = blockIdx.x * blockDim.x + threadIdx.x;
  if (i < n) p[i] = 0.0f;
}

__global__ void k_lens(const int* __restrict__ text, float* __restrict__ out_lens,
                       int* __restrict__ lens_i) {
  __shared__ int s;
  const int b = blockIdx.x, tid = threadIdx.x;
  if (tid == 0) s = 0;
  __syncthreads();
  int c = 0;
  for (int tt = tid; tt < kT; tt += 256) c += (text[b * kT + tt] != 0);
  atomicAdd(&s, c);
  __syncthreads();
  if (tid == 0) { out_lens[b] = (float)s; lens_i[b] = s; }
}

// x[row, 0:608] = concat(word_emb[text], sent_emb[sent], zero-pad) as bf16 (row-major, Kp=608)
__global__ void k_embed(const int* __restrict__ text, const int* __restrict__ sent,
                        const float* __restrict__ wemb, const float* __restrict__ semb,
                        bf16_t* __restrict__ xbf) {
  const int row = blockIdx.x;
  const int wid = text[row], sid = sent[row];
  for (int d = threadIdx.x; d < kDp; d += blockDim.x) {
    float v = 0.0f;
    if (d < kE)      v = wemb[(size_t)wid * kE + d];
    else if (d < kD) v = semb[(size_t)sid * kE + (d - kE)];
    xbf[(size_t)row * kDp + d] = (bf16_t)v;
  }
}

// src (K x N, f32, row-major) -> dst (N x Kp, bf16, column-of-src contiguous; K-pad zeros)
__global__ void k_transpose_pad(const float* __restrict__ src, bf16_t* __restrict__ dst,
                                int K, int N, int Kp) {
  const int n = blockIdx.x;
  for (int k = threadIdx.x; k < Kp; k += blockDim.x) {
    float v = (k < K) ? src[(size_t)k * N + n] : 0.0f;
    dst[(size_t)n * Kp + k] = (bf16_t)v;
  }
}

// ---------------- input projection: xz = x @ Wk + b  (M=16384, K=608, N=2048, both dirs)
// Whole B panel (64 cols x 608 K, 76KB bf16) DMA'd into LDS once by the TDM;
// K-loop then runs barrier-free: global A loads + ds B loads + wmma.
__global__ void __launch_bounds__(256) k_gemm_xz(
    const bf16_t* __restrict__ x, const bf16_t* __restrict__ wkT_f,
    const bf16_t* __restrict__ wkT_b, const float* __restrict__ bias_f,
    const float* __restrict__ bias_b, bf16_t* __restrict__ xz_f, bf16_t* __restrict__ xz_b) {
  const int dir = blockIdx.z;
  const bf16_t* wkT = dir ? wkT_b : wkT_f;
  const float* bias = dir ? bias_b : bias_f;
  bf16_t* xz = dir ? xz_b : xz_f;

  const int tid = threadIdx.x;
  const int w = tid >> 5, lane = tid & 31;
  const int lhalf = lane >> 4, lmod = lane & 15;
  const int m0 = blockIdx.x * 128 + w * 16;  // 8 waves stacked in M
  const int n0 = blockIdx.y * 64;

  __shared__ bf16_t lds_b[64][kDp];  // full B panel: [n][k], 77824 B

#if USE_TDM
  if (tid < 32) {  // wave 0 issues the DMA; TENSORcnt is per-wave
    tdm_load_2d_bf16(wkT + (size_t)n0 * kDp, &lds_b[0][0],
                     /*tile_d0=*/kDp, /*tile_d1=*/64,
                     /*tensor_d0=*/kDp, /*tensor_d1=*/kG,
                     /*stride0=*/(unsigned long long)kDp);
    __builtin_amdgcn_s_wait_tensorcnt(0);
  }
#else
  {
    constexpr int chunks = 64 * kDp * 2 / 16;  // 16B chunks
    constexpr int per_row = kDp * 2 / 16;      // 76
    for (int c = tid; c < chunks; c += 256) {
      const int row = c / per_row, seg = c % per_row;
      *(uint4*)(&lds_b[row][seg * 8]) =
          *(const uint4*)(wkT + (size_t)(n0 + row) * kDp + seg * 8);
    }
  }
#endif
  __syncthreads();

  v8f acc[4] = {};
  for (int k0 = 0; k0 < kDp; k0 += 32) {
    // A fragment: elem e -> K = 16*lhalf + e (matches B)
    v16bf a;
    {
      const bf16_t* p = x + (size_t)(m0 + lmod) * kDp + k0 + lhalf * 16;
      if (k0 + 32 < kDp) __builtin_prefetch(p + 32, 0, 1);
#pragma unroll
      for (int i = 0; i < 16; ++i) a[i] = p[i];
    }
#pragma unroll
    for (int j = 0; j < 4; ++j) {
      // B fragment: lanes 0-15 K=0..15, lanes 16-31 K=16..31, elem i -> K = 16*lhalf + i
      v16bf bfrag;
      const bf16_t* q = &lds_b[j * 16 + lmod][k0 + lhalf * 16];
#pragma unroll
      for (int i = 0; i < 16; ++i) bfrag[i] = q[i];
      acc[j] = wmma_bf16(a, bfrag, acc[j]);
    }
  }

#pragma unroll
  for (int j = 0; j < 4; ++j) {
#pragma unroll
    for (int r = 0; r < 8; ++r) {  // D elem r -> row m0 + r + 8*lhalf, col = lmod
      const int m = m0 + r + 8 * lhalf;
      const int n = n0 + j * 16 + lmod;
      xz[(size_t)m * kG + n] = (bf16_t)(acc[j][r] + bias[n]);
    }
  }
}

// ---------------- one recurrence step (both directions), fused GEMM + gates
// grid (32 hcol-tiles, 2 dirs), 256 threads. Whole h_{t-1} panel (64x512 bf16,
// 64KB) DMA'd into LDS once via TDM -> barrier-free 16-iteration K-loop on the
// serial critical path.
__global__ void __launch_bounds__(256) k_step(
    int t, const bf16_t* __restrict__ xz_f, const bf16_t* __restrict__ xz_b,
    const bf16_t* __restrict__ wrT_f, const bf16_t* __restrict__ wrT_b,
    bf16_t* __restrict__ h_all, float* __restrict__ c_state) {
  const int dir = blockIdx.y;
  const bf16_t* xz = dir ? xz_b : xz_f;
  const bf16_t* wrT = dir ? wrT_b : wrT_f;
  const int time_idx = dir ? (kT - 1 - t) : t;
  const int prev_time = dir ? (kT - t) : (t - 1);  // only used when t>0
  const int hcol0 = blockIdx.x * 16;

  const int tid = threadIdx.x;
  const int w = tid >> 5, lane = tid & 31;
  const int lhalf = lane >> 4, lmod = lane & 15;
  const int mtile = w & 3, gbase = (w >> 2) * 2;  // each wave: 1 m-tile x 2 gates

  __shared__ bf16_t lds_a[64][kH];     // h_{t-1} panel, 64KB
  __shared__ float lds_z[4][64][16];   // gate pre-activations, 16KB

  v8f acc[2] = {};

  if (t > 0) {
#if USE_TDM
    if (tid < 32) {
      tdm_load_2d_bf16(h_all + (size_t)prev_time * (2 * kH) + dir * kH, &lds_a[0][0],
                       /*tile_d0=*/kH, /*tile_d1=*/kB,
                       /*tensor_d0=*/kH, /*tensor_d1=*/kB,
                       /*stride0=*/(unsigned long long)kT * (2 * kH));
      __builtin_amdgcn_s_wait_tensorcnt(0);
    }
#else
    {
      constexpr int chunks = kB * kH * 2 / 16;
      constexpr int per_row = kH * 2 / 16;  // 64
      for (int c = tid; c < chunks; c += 256) {
        const int row = c / per_row, seg = c % per_row;
        *(uint4*)(&lds_a[row][seg * 8]) =
            *(const uint4*)(h_all + ((size_t)row * kT + prev_time) * (2 * kH) + dir * kH + seg * 8);
      }
    }
#endif
    __syncthreads();

    for (int k0 = 0; k0 < kH; k0 += 32) {
      v16bf a;
      {
        const bf16_t* p = &lds_a[mtile * 16 + lmod][k0 + lhalf * 16];
#pragma unroll
        for (int i = 0; i < 16; ++i) a[i] = p[i];
      }
#pragma unroll
      for (int gi = 0; gi < 2; ++gi) {
        const int col = (gbase + gi) * kH + hcol0 + lmod;
        const bf16_t* q = wrT + (size_t)col * kH + k0 + lhalf * 16;  // contiguous 32B
        v16bf bfrag;
#pragma unroll
        for (int i = 0; i < 16; ++i) bfrag[i] = q[i];
        acc[gi] = wmma_bf16(a, bfrag, acc[gi]);
      }
    }
  }

  // z = h@Wr + xz ; scatter to LDS for gate fusion
#pragma unroll
  for (int gi = 0; gi < 2; ++gi) {
    const int g = gbase + gi;
#pragma unroll
    for (int r = 0; r < 8; ++r) {
      const int m = mtile * 16 + r + 8 * lhalf;
      const float xv = (float)xz[((size_t)m * kT + time_idx) * kG + g * kH + hcol0 + lmod];
      lds_z[g][m][lmod] = acc[gi][r] + xv;
    }
  }
  __syncthreads();

  // elementwise LSTM cell: 64 batch x 16 cols = 1024 items
#pragma unroll
  for (int ii = 0; ii < 4; ++ii) {
    const int idx = tid + 256 * ii;
    const int b = idx >> 4, j = idx & 15;
    const float zi = lds_z[0][b][j], zf = lds_z[1][b][j];
    const float zg = lds_z[2][b][j], zo = lds_z[3][b][j];
    const int col = hcol0 + j;
    const size_t cidx = ((size_t)dir * kB + b) * kH + col;  // block-private columns
    float c = sigmoidf_(zf) * c_state[cidx] + sigmoidf_(zi) * tanhf(zg);
    float h = sigmoidf_(zo) * tanhf(c);
    c_state[cidx] = c;
    h_all[((size_t)b * kT + time_idx) * (2 * kH) + dir * kH + col] = (bf16_t)h;
  }
}

// ---------------- logits = h @ Wd + bd (N=9: one wave per row, shfl reduce)
__global__ void __launch_bounds__(256) k_logits(const bf16_t* __restrict__ h_all,
                                                const float* __restrict__ Wd,
                                                const float* __restrict__ bd,
                                                float* __restrict__ out) {
  const int w = threadIdx.x >> 5, lane = threadIdx.x & 31;
  const int row = blockIdx.x * 8 + w;
  float acc[kL];
#pragma unroll
  for (int l = 0; l < kL; ++l) acc[l] = 0.0f;
  const bf16_t* hrow = h_all + (size_t)row * (2 * kH);
  for (int j = lane; j < 2 * kH; j += 32) {
    const float hv = (float)hrow[j];
    const float* wr = Wd + (size_t)j * kL;
#pragma unroll
    for (int l = 0; l < kL; ++l) acc[l] += hv * wr[l];
  }
#pragma unroll
  for (int l = 0; l < kL; ++l) acc[l] = wave_sum(acc[l]);
  if (lane == 0) {
#pragma unroll
    for (int l = 0; l < kL; ++l) out[(size_t)row * kL + l] = acc[l] + bd[l];
  }
}

// ---------------- CRF log-likelihood: one wave (32) per batch row, lanes = states
__global__ void k_crf_ll(const float* __restrict__ logits, const int* __restrict__ labels,
                         const float* __restrict__ trans, const int* __restrict__ lens_i,
                         float* __restrict__ ll_out) {
  const int b = blockIdx.x, lane = threadIdx.x;
  const int len = lens_i[b];
  const float* lg = logits + (size_t)b * kT * kL;
  __shared__ float salpha[kL];

  float tcol[kL];
  if (lane < kL) {
#pragma unroll
    for (int i = 0; i < kL; ++i) tcol[i] = trans[i * kL + lane];
    salpha[lane] = lg[lane];
  }
  float up = 0.0f, bp = 0.0f;
  for (int tt = lane; tt < kT; tt += 32) {
    const int lab = labels[b * kT + tt];
    const float mf = (tt < len) ? 1.0f : 0.0f;
    up += lg[tt * kL + lab] * mf;
    if (tt >= 1) bp += trans[labels[b * kT + tt - 1] * kL + lab] * mf;
  }
  up = wave_sum(up);
  bp = wave_sum(bp);
  __syncthreads();

  for (int tt = 1; tt < kT; ++tt) {
    float al[kL];
#pragma unroll
    for (int i = 0; i < kL; ++i) al[i] = salpha[i];
    float nv = 0.0f;
    if (lane < kL) {
      float m = -1e30f;
#pragma unroll
      for (int i = 0; i < kL; ++i) m = fmaxf(m, al[i] + tcol[i]);
      float s = 0.0f;
#pragma unroll
      for (int i = 0; i < kL; ++i) s += expf(al[i] + tcol[i] - m);
      nv = m + logf(s) + lg[tt * kL + lane];
    }
    __syncthreads();
    if (lane < kL && tt < len) salpha[lane] = nv;
    __syncthreads();
  }
  if (lane == 0) {
    float m = -1e30f;
    for (int i = 0; i < kL; ++i) m = fmaxf(m, salpha[i]);
    float s = 0.0f;
    for (int i = 0; i < kL; ++i) s += expf(salpha[i] - m);
    ll_out[b] = up + bp - (m + logf(s));
  }
}

// ---------------- CRF Viterbi decode
__global__ void k_decode(const float* __restrict__ logits, const float* __restrict__ trans,
                         const int* __restrict__ lens_i, int* __restrict__ bp_ws,
                         float* __restrict__ tags_out) {
  const int b = blockIdx.x, lane = threadIdx.x;
  const int len = lens_i[b];
  const float* lg = logits + (size_t)b * kT * kL;
  int* bp = bp_ws + (size_t)b * kT * kL;
  __shared__ float salpha[kL];

  float tcol[kL];
  if (lane < kL) {
#pragma unroll
    for (int i = 0; i < kL; ++i) tcol[i] = trans[i * kL + lane];
    salpha[lane] = lg[lane];
  }
  __syncthreads();

  for (int tt = 1; tt < kT; ++tt) {
    float al[kL];
#pragma unroll
    for (int i = 0; i < kL; ++i) al[i] = salpha[i];
    int best = 0;
    float bv = 0.0f;
    if (lane < kL) {
      bv = al[0] + tcol[0];
#pragma unroll
      for (int i = 1; i < kL; ++i) {
        const float v = al[i] + tcol[i];
        if (v > bv) { bv = v; best = i; }  // ties -> first index (matches argmax)
      }
    }
    __syncthreads();
    if (lane < kL) {
      if (tt < len) { salpha[lane] = bv + lg[tt * kL + lane]; bp[tt * kL + lane] = best; }
      else          bp[tt * kL + lane] = lane;  // identity for masked steps
    }
    __syncthreads();
  }
  if (lane == 0) {
    int last = 0;
    float bv = salpha[0];
    for (int j = 1; j < kL; ++j)
      if (salpha[j] > bv) { bv = salpha[j]; last = j; }
    float* tout = tags_out + (size_t)b * kT;
    tout[kT - 1] = (float)last;
    int tag = last;
    for (int tt = kT - 1; tt >= 1; --tt) {
      tag = bp[tt * kL + tag];
      tout[tt - 1] = (float)tag;
    }
  }
}

// ---------------- host ----------------
extern "C" void kernel_launch(void* const* d_in, const int* in_sizes, int n_in,
                              void* d_out, int out_size, void* d_ws, size_t ws_size,
                              hipStream_t stream) {
  (void)in_sizes; (void)n_in; (void)out_size; (void)ws_size;
  const int*   text   = (const int*)d_in[0];
  const int*   sent   = (const int*)d_in[1];
  const int*   labels = (const int*)d_in[2];
  const float* wemb   = (const float*)d_in[3];
  const float* semb   = (const float*)d_in[4];
  const float* Wk_f   = (const float*)d_in[5];
  const float* Wr_f   = (const float*)d_in[6];
  const float* b_f    = (const float*)d_in[7];
  const float* Wk_b   = (const float*)d_in[8];
  const float* Wr_b   = (const float*)d_in[9];
  const float* b_b    = (const float*)d_in[10];
  const float* Wd     = (const float*)d_in[11];
  const float* bd     = (const float*)d_in[12];
  const float* trans  = (const float*)d_in[13];
  float* out = (float*)d_out;

  char* ws = (char*)d_ws;
  size_t off = 0;
  auto alloc = [&](size_t bytes) -> char* {
    char* p = ws + off;
    off += (bytes + 255) & ~(size_t)255;
    return p;
  };
  bf16_t* x_bf    = (bf16_t*)alloc((size_t)kB * kT * kDp * 2);   // ~20 MB
  bf16_t* wkT_f   = (bf16_t*)alloc((size_t)kG * kDp * 2);
  bf16_t* wkT_b   = (bf16_t*)alloc((size_t)kG * kDp * 2);
  bf16_t* wrT_f   = (bf16_t*)alloc((size_t)kG * kH * 2);
  bf16_t* wrT_b   = (bf16_t*)alloc((size_t)kG * kH * 2);
  bf16_t* xz_f    = (bf16_t*)alloc((size_t)kB * kT * kG * 2);    // 67 MB
  bf16_t* xz_b    = (bf16_t*)alloc((size_t)kB * kT * kG * 2);    // 67 MB
  bf16_t* h_all   = (bf16_t*)alloc((size_t)kB * kT * 2 * kH * 2);// 34 MB
  float*  c_state = (float*)alloc((size_t)2 * kB * kH * 4);
  int*    lens_i  = (int*)alloc((size_t)kB * 4);
  int*    bp_ws   = (int*)alloc((size_t)kB * kT * kL * 4);

  k_zero_f32<<<dim3((2 * kB * kH + 255) / 256), 256, 0, stream>>>(c_state, 2 * kB * kH);
  k_lens<<<kB, 256, 0, stream>>>(text, out + LENS_OFF, lens_i);
  k_embed<<<kB * kT, 256, 0, stream>>>(text, sent, wemb, semb, x_bf);
  k_transpose_pad<<<kG, 256, 0, stream>>>(Wk_f, wkT_f, kD, kG, kDp);
  k_transpose_pad<<<kG, 256, 0, stream>>>(Wk_b, wkT_b, kD, kG, kDp);
  k_transpose_pad<<<kG, 256, 0, stream>>>(Wr_f, wrT_f, kH, kG, kH);
  k_transpose_pad<<<kG, 256, 0, stream>>>(Wr_b, wrT_b, kH, kG, kH);

  k_gemm_xz<<<dim3(kB * kT / 128, kG / 64, 2), 256, 0, stream>>>(
      x_bf, wkT_f, wkT_b, b_f, b_b, xz_f, xz_b);

  for (int t = 0; t < kT; ++t)
    k_step<<<dim3(kH / 16, 2), 256, 0, stream>>>(t, xz_f, xz_b, wrT_f, wrT_b, h_all, c_state);

  k_logits<<<kB * kT / 8, 256, 0, stream>>>(h_all, Wd, bd, out + LOGITS_OFF);
  k_crf_ll<<<kB, 32, 0, stream>>>(out + LOGITS_OFF, labels, trans, lens_i, out + LL_OFF);
  k_decode<<<kB, 32, 0, stream>>>(out + LOGITS_OFF, trans, lens_i, bp_ws, out + TAGS_OFF);
}
